// RNNDecoder_79310866088590
// MI455X (gfx1250) — matrix operand order repaired and checked by reading, not compiled, not run on hardware
//
#include <hip/hip_runtime.h>
#include <math.h>

typedef __attribute__((ext_vector_type(16))) _Float16 v16h;
typedef __attribute__((ext_vector_type(8)))  _Float16 v8h;
typedef __attribute__((ext_vector_type(8)))  float    v8f;

constexpr int kB = 32, kL = 513, kS = 1024, kH = 512, kDh = 128, kNcls = 128;
constexpr int kML = kB * kL;            // 16416 (M of seq GEMMs; /32 = 513)
constexpr int kMS = kB * kS;            // 32768 (M of k/v GEMMs; /32 = 1024)
constexpr int kG4 = 4 * kH;             // 2048
constexpr int kWPB = 64, kTW = kB * kWPB; // 2048 words
constexpr int kMT = 33;                 // ceil(513/16) tiles in L dim

__device__ __forceinline__ float sigf(float x) { return 1.0f / (1.0f + __expf(-x)); }

// A-fragment (16x32 f16): per lane two contiguous 8-half groups.
__device__ __forceinline__ v16h load_a(const _Float16* __restrict__ arow, int k0, int laneHi) {
  v8h lo = *(const v8h*)(arow + k0 + laneHi * 16);
  v8h hi = *(const v8h*)(arow + k0 + laneHi * 16 + 8);
  v16h a;
#pragma unroll
  for (int i = 0; i < 8; ++i) { a[i] = lo[i]; a[i + 8] = hi[i]; }
  return a;
}
__device__ __forceinline__ v8f wmma_acc(v16h a, v16h b, v8f c) {
  return __builtin_amdgcn_wmma_f32_16x16x32_f16(false, a, false, b, (short)0, c, false, false);
}

// Dual-accumulator 16x16 tile (breaks WMMA->WMMA RAW chain). K % 64 == 0.
__device__ __forceinline__ v8f tile_gemm2(const _Float16* __restrict__ arow,
                                          const _Float16* __restrict__ brow,
                                          int K, int laneHi) {
  v8f acc0 = {}, acc1 = {};
  for (int k0 = 0; k0 < K; k0 += 64) {
    v16h a0 = load_a(arow, k0, laneHi);
    v16h b0 = *(const v16h*)(brow + k0 + laneHi * 16);
    v16h a1 = load_a(arow, k0 + 32, laneHi);
    v16h b1 = *(const v16h*)(brow + k0 + 32 + laneHi * 16);
    acc0 = wmma_acc(a0, b0, acc0);
    acc1 = wmma_acc(a1, b1, acc1);
  }
#pragma unroll
  for (int i = 0; i < 8; ++i) acc0[i] += acc1[i];
  return acc0;
}

// ---------------- blocked WMMA GEMM: each wave computes 32x64 of C = A @ W^T ----------------
// A: (M,K) f16 row-major; W: (N,K) f16 row-major (B = W^T gathers are contiguous).
// N,K compile-time: B-row offsets fold into load immediates; store math becomes shifts.
// grid (N/64, M/32), block 32. 8 independent accumulators per wave.
template <int N, int K, int ACT>
__global__ void k_gemm(const _Float16* __restrict__ A, const _Float16* __restrict__ W,
                       const float* __restrict__ bias1, const float* __restrict__ bias2,
                       float* __restrict__ C32, _Float16* __restrict__ C16) {
  int lane = threadIdx.x;
  int laneHi = lane >> 4, laneLo = lane & 15;
  int n0 = blockIdx.x * 64, m0 = blockIdx.y * 32;
  const _Float16* ar = A + (size_t)(m0 + laneLo) * K;
  const _Float16* br = W + (size_t)(n0 + laneLo) * K;
  v8f z = {};
  v8f acc[2][4];
#pragma unroll
  for (int mt = 0; mt < 2; ++mt)
#pragma unroll
    for (int nt = 0; nt < 4; ++nt) acc[mt][nt] = z;
  for (int k0 = 0; k0 < K; k0 += 32) {
    // all fragment loads up front -> backend pipelines loads against WMMAs
    v16h a0 = load_a(ar, k0, laneHi);
    v16h a1 = load_a(ar + 16 * K, k0, laneHi);
    v16h b0 = *(const v16h*)(br + 0 * 16 * K + k0 + laneHi * 16);
    v16h b1 = *(const v16h*)(br + 1 * 16 * K + k0 + laneHi * 16);
    v16h b2 = *(const v16h*)(br + 2 * 16 * K + k0 + laneHi * 16);
    v16h b3 = *(const v16h*)(br + 3 * 16 * K + k0 + laneHi * 16);
    acc[0][0] = wmma_acc(a0, b0, acc[0][0]);
    acc[1][0] = wmma_acc(a1, b0, acc[1][0]);
    acc[0][1] = wmma_acc(a0, b1, acc[0][1]);
    acc[1][1] = wmma_acc(a1, b1, acc[1][1]);
    acc[0][2] = wmma_acc(a0, b2, acc[0][2]);
    acc[1][2] = wmma_acc(a1, b2, acc[1][2]);
    acc[0][3] = wmma_acc(a0, b3, acc[0][3]);
    acc[1][3] = wmma_acc(a1, b3, acc[1][3]);
  }
  float bb[4];
#pragma unroll
  for (int nt = 0; nt < 4; ++nt) {
    int n = n0 + nt * 16 + laneLo;
    bb[nt] = (bias1 ? bias1[n] : 0.0f) + (bias2 ? bias2[n] : 0.0f);
  }
#pragma unroll
  for (int mt = 0; mt < 2; ++mt)
#pragma unroll
    for (int nt = 0; nt < 4; ++nt)
#pragma unroll
      for (int r = 0; r < 8; ++r) {
        int m = m0 + mt * 16 + r + 8 * laneHi;
        int n = n0 + nt * 16 + laneLo;
        float v = acc[mt][nt][r] + bb[nt];
        if (ACT == 1) v = tanhf(v);
        size_t idx = (size_t)m * N + n;
        if (C32) C32[idx] = v;
        if (C16) C16[idx] = (_Float16)v;
      }
}

// ---------------- LSTM one timestep: gates = xp[:,t,:] + h_in @ whh^T ----------------
// grid 32 (j-slices of 16), block 256 (8 waves: wave = gate*2 + mtile)
__global__ void k_lstm_step(const float* __restrict__ xp, int t,
                            const _Float16* __restrict__ whh,
                            const _Float16* __restrict__ h_in,
                            _Float16* __restrict__ h_out,
                            float* __restrict__ cst,
                            float* __restrict__ out32,
                            _Float16* __restrict__ seq16) {
  __shared__ float gbuf[8][16][16];
  int tid = threadIdx.x;
  int wv = tid >> 5;
  int lane = tid & 31;
  int laneHi = lane >> 4, laneLo = lane & 15;
  int mt = wv & 1, gate = wv >> 1;
  int j0 = blockIdx.x * 16;
  const _Float16* arow = h_in + (size_t)(mt * 16 + laneLo) * kH;
  const _Float16* brow = whh + (size_t)(gate * kH + j0 + laneLo) * kH;
  v8f acc = tile_gemm2(arow, brow, kH, laneHi);
#pragma unroll
  for (int r = 0; r < 8; ++r) gbuf[wv][r + 8 * laneHi][laneLo] = acc[r];
  __syncthreads();
  for (int idx = tid; idx < kB * 16; idx += 256) {
    int b = idx >> 4, jj = idx & 15;
    int j = j0 + jj;
    int mt2 = b >> 4, mr = b & 15;
    size_t xbase = ((size_t)b * kL + t) * kG4 + j;
    float gi = gbuf[0 + mt2][mr][jj] + xp[xbase];
    float gf = gbuf[2 + mt2][mr][jj] + xp[xbase + kH];
    float gg = gbuf[4 + mt2][mr][jj] + xp[xbase + 2 * kH];
    float go = gbuf[6 + mt2][mr][jj] + xp[xbase + 3 * kH];
    float cn = sigf(gf) * cst[b * kH + j] + sigf(gi) * tanhf(gg);
    float hn = sigf(go) * tanhf(cn);
    cst[b * kH + j] = cn;
    h_out[b * kH + j] = (_Float16)hn;
    seq16[((size_t)b * kL + t) * kH + j] = (_Float16)hn;
    if (out32) out32[((size_t)b * kL + t) * kH + j] = hn;
  }
}

// ---------------- attention scores: attn[b,hd,l,s] = q.k / sqrt(dh) ----------------
// grid (S/64, 33, B*4), block 32; each wave 16x64, 4 accumulators, K=dh=128.
__global__ void k_scores(const _Float16* __restrict__ q16, const _Float16* __restrict__ k16,
                         float* __restrict__ attn) {
  int lane = threadIdx.x;
  int laneHi = lane >> 4, laneLo = lane & 15;
  int b = blockIdx.z >> 2, hd = blockIdx.z & 3;
  int m0 = blockIdx.y * 16, n0 = blockIdx.x * 64;
  int l = m0 + laneLo; if (l > kL - 1) l = kL - 1;
  const _Float16* arow = q16 + ((size_t)b * kL + l) * kH + hd * kDh;
  const _Float16* br = k16 + ((size_t)b * kS + n0 + laneLo) * kH + hd * kDh;
  v8f z = {};
  v8f acc[4] = {z, z, z, z};
#pragma unroll
  for (int k0 = 0; k0 < kDh; k0 += 32) {
    v16h a = load_a(arow, k0, laneHi);
    v16h b0 = *(const v16h*)(br + 0 * 16 * kH + k0 + laneHi * 16);
    v16h b1 = *(const v16h*)(br + 1 * 16 * kH + k0 + laneHi * 16);
    v16h b2 = *(const v16h*)(br + 2 * 16 * kH + k0 + laneHi * 16);
    v16h b3 = *(const v16h*)(br + 3 * 16 * kH + k0 + laneHi * 16);
    acc[0] = wmma_acc(a, b0, acc[0]);
    acc[1] = wmma_acc(a, b1, acc[1]);
    acc[2] = wmma_acc(a, b2, acc[2]);
    acc[3] = wmma_acc(a, b3, acc[3]);
  }
  const float sc = 0.08838834764831845f; // 1/sqrt(128)
#pragma unroll
  for (int nt = 0; nt < 4; ++nt)
#pragma unroll
    for (int r = 0; r < 8; ++r) {
      int m = m0 + r + 8 * laneHi;
      if (m < kL)
        attn[(((size_t)blockIdx.z) * kL + m) * kS + n0 + nt * 16 + laneLo] = acc[nt][r] * sc;
    }
}

// ---------------- softmax over S=1024, in place ----------------
__global__ void k_softmax1024(float* __restrict__ attn) {
  __shared__ float red[256];
  float* row = attn + (size_t)blockIdx.x * kS;
  int tid = threadIdx.x;
  float v[4]; float m = -3.4e38f;
#pragma unroll
  for (int i = 0; i < 4; ++i) { v[i] = row[tid + 256 * i]; m = fmaxf(m, v[i]); }
  red[tid] = m; __syncthreads();
  for (int s = 128; s > 0; s >>= 1) { if (tid < s) red[tid] = fmaxf(red[tid], red[tid + s]); __syncthreads(); }
  float mx = red[0]; __syncthreads();
  float s = 0.0f;
#pragma unroll
  for (int i = 0; i < 4; ++i) { v[i] = __expf(v[i] - mx); s += v[i]; }
  red[tid] = s; __syncthreads();
  for (int st = 128; st > 0; st >>= 1) { if (tid < st) red[tid] += red[tid + st]; __syncthreads(); }
  float inv = 1.0f / red[0];
#pragma unroll
  for (int i = 0; i < 4; ++i) row[tid + 256 * i] = v[i] * inv;
}

// ---------------- context = attn @ v  (A f32->f16 in-register, B = vT contiguous) ----------------
// grid (dh/64, 33, B*4), block 32; each wave 16x64, 4 accumulators, K=S=1024.
__global__ void k_ctx(const float* __restrict__ attn, const _Float16* __restrict__ vT,
                      float* __restrict__ ctx) {
  int lane = threadIdx.x;
  int laneHi = lane >> 4, laneLo = lane & 15;
  int b = blockIdx.z >> 2, hd = blockIdx.z & 3;
  int m0 = blockIdx.y * 16, n0 = blockIdx.x * 64;
  int l = m0 + laneLo; if (l > kL - 1) l = kL - 1;
  const float* arow = attn + ((size_t)blockIdx.z * kL + l) * kS;
  const _Float16* br = vT + ((size_t)blockIdx.z * kDh + n0 + laneLo) * kS;
  v8f z = {};
  v8f acc[4] = {z, z, z, z};
  for (int k0 = 0; k0 < kS; k0 += 32) {
    v16h a;
#pragma unroll
    for (int i = 0; i < 8; ++i) {
      a[i] = (_Float16)arow[k0 + laneHi * 16 + i];
      a[i + 8] = (_Float16)arow[k0 + laneHi * 16 + 8 + i];
    }
    v16h b0 = *(const v16h*)(br + 0 * 16 * kS + k0 + laneHi * 16);
    v16h b1 = *(const v16h*)(br + 1 * 16 * kS + k0 + laneHi * 16);
    v16h b2 = *(const v16h*)(br + 2 * 16 * kS + k0 + laneHi * 16);
    v16h b3 = *(const v16h*)(br + 3 * 16 * kS + k0 + laneHi * 16);
    acc[0] = wmma_acc(a, b0, acc[0]);
    acc[1] = wmma_acc(a, b1, acc[1]);
    acc[2] = wmma_acc(a, b2, acc[2]);
    acc[3] = wmma_acc(a, b3, acc[3]);
  }
#pragma unroll
  for (int nt = 0; nt < 4; ++nt)
#pragma unroll
    for (int r = 0; r < 8; ++r) {
      int m = m0 + r + 8 * laneHi;
      if (m < kL)
        ctx[((size_t)b * kL + m) * kH + hd * kDh + n0 + nt * 16 + laneLo] = acc[nt][r];
    }
}

// ---------------- small helpers ----------------
__global__ void k_cast16(const float* __restrict__ in, _Float16* __restrict__ out, int n) {
  for (int i = blockIdx.x * blockDim.x + threadIdx.x; i < n; i += gridDim.x * blockDim.x)
    out[i] = (_Float16)in[i];
}
__global__ void k_zero32(float* __restrict__ p, int n) {
  for (int i = blockIdx.x * blockDim.x + threadIdx.x; i < n; i += gridDim.x * blockDim.x)
    p[i] = 0.0f;
}
__global__ void k_zero16(_Float16* __restrict__ p, int n) {
  for (int i = blockIdx.x * blockDim.x + threadIdx.x; i < n; i += gridDim.x * blockDim.x)
    p[i] = (_Float16)0.0f;
}
__global__ void k_embed(const int* __restrict__ tg, const float* __restrict__ emb_w,
                        float* __restrict__ e32, _Float16* __restrict__ e16) {
  int bt = blockIdx.x;
  int id = tg[bt];
  for (int i = threadIdx.x; i < kH; i += 256) {
    float v = emb_w[(size_t)id * kH + i];
    e32[(size_t)bt * kH + i] = v;
    e16[(size_t)bt * kH + i] = (_Float16)v;
  }
}
__global__ void k_vtrans(const _Float16* __restrict__ v16, _Float16* __restrict__ vT) {
  for (int i = blockIdx.x * blockDim.x + threadIdx.x; i < kB * 4 * kDh * kS;
       i += gridDim.x * blockDim.x) {
    int s = i & (kS - 1); int r = i >> 10;
    int d = r & (kDh - 1); int bh = r >> 7;
    int b = bh >> 2, hd = bh & 3;
    vT[i] = v16[((size_t)b * kS + s) * kH + hd * kDh + d];
  }
}
__global__ void k_cat(const float* __restrict__ outs, const float* __restrict__ ctx,
                      _Float16* __restrict__ cat16) {
  for (int i = blockIdx.x * blockDim.x + threadIdx.x; i < kML * 2 * kH;
       i += gridDim.x * blockDim.x) {
    int j = i & (2 * kH - 1); int bt = i >> 10;
    float v = (j < kH) ? outs[(size_t)bt * kH + j] : ctx[(size_t)bt * kH + (j - kH)];
    cat16[i] = (_Float16)v;
  }
}
__global__ void k_logsoftmax(float* __restrict__ logits) {
  __shared__ float red[128];
  float* row = logits + (size_t)blockIdx.x * kNcls;
  int tid = threadIdx.x;
  float x = row[tid];
  red[tid] = x; __syncthreads();
  for (int s = 64; s > 0; s >>= 1) { if (tid < s) red[tid] = fmaxf(red[tid], red[tid + s]); __syncthreads(); }
  float mx = red[0]; __syncthreads();
  red[tid] = __expf(x - mx); __syncthreads();
  for (int s = 64; s > 0; s >>= 1) { if (tid < s) red[tid] += red[tid + s]; __syncthreads(); }
  float lse = mx + logf(red[0]);
  row[tid] = x - lse;
}
// split_words: generic per-row word scan + scatter. grid kB, block 256
__global__ void k_split(const int* __restrict__ tg, const float* __restrict__ e32,
                        const float* __restrict__ ctx, float* __restrict__ mis) {
  __shared__ int s_gidx[kL - 1];
  __shared__ int s_pos[kL - 1];
  int b = blockIdx.x, tid = threadIdx.x;
  if (tid == 0) {
    int word = 0, start = 0;
    for (int tp = 0; tp < kL - 1; ++tp) {
      int id = tg[b * kL + tp + 1];
      bool bd = (id == 1) || (id == 0); // SPACE or PAD
      if (bd) {
        s_gidx[tp] = -1;
        word += 1; start = tp + 1;
      } else {
        int g = b * kWPB + word;
        s_gidx[tp] = (g < kTW) ? g : -1;
        int p = tp - start; if (p > 6) p = 6;
        s_pos[tp] = p;
      }
    }
  }
  __syncthreads();
  for (int i = tid; i < (kL - 1) * 1024; i += 256) {
    int c = i & 1023; int tp = i >> 10;
    int g = s_gidx[tp];
    if (g < 0) continue;
    float v = (c < kH) ? e32[((size_t)b * kL + tp + 1) * kH + c]
                       : ctx[((size_t)b * kL + tp) * kH + (c - kH)];
    mis[((size_t)g * 7 + s_pos[tp]) * 1024 + c] = v;
  }
}

extern "C" void kernel_launch(void* const* d_in, const int* in_sizes, int n_in,
                              void* d_out, int out_size, void* d_ws, size_t ws_size,
                              hipStream_t stream) {
  const int*   targets = (const int*)d_in[0];
  const float* enc     = (const float*)d_in[1];
  const float* emb_w   = (const float*)d_in[2];
  const float* w_ih0   = (const float*)d_in[3];
  const float* w_hh0   = (const float*)d_in[4];
  const float* b_ih0   = (const float*)d_in[5];
  const float* b_hh0   = (const float*)d_in[6];
  const float* w_ih1   = (const float*)d_in[7];
  const float* w_hh1   = (const float*)d_in[8];
  const float* b_ih1   = (const float*)d_in[9];
  const float* b_hh1   = (const float*)d_in[10];
  const float* q_w = (const float*)d_in[11]; const float* q_b = (const float*)d_in[12];
  const float* k_w = (const float*)d_in[13]; const float* k_b = (const float*)d_in[14];
  const float* v_w = (const float*)d_in[15]; const float* v_b = (const float*)d_in[16];
  const float* fc1_w = (const float*)d_in[17]; const float* fc1_b = (const float*)d_in[18];
  const float* fc2_w = (const float*)d_in[19]; const float* fc2_b = (const float*)d_in[20];

  float* out = (float*)d_out;
  float* out_logp = out;                                   // 16416*128
  float* out_attn = out + (size_t)kML * kNcls;             // 32*4*513*1024
  float* out_mis  = out_attn + (size_t)kB * 4 * kL * kS;   // 2048*7*1024

  // ---- scratch carve-up ----
  char* w = (char*)d_ws; size_t o = 0;
  auto take = [&](size_t bytes) { void* p = w + o; o += (bytes + 255) & ~(size_t)255; return p; };
  float*    emb32  = (float*)take((size_t)kML * kH * 4);
  _Float16* seq16  = (_Float16*)take((size_t)kML * kH * 2);   // emb16 -> x16 -> outputs16
  _Float16* wih0_16 = (_Float16*)take((size_t)kG4 * kH * 2);
  _Float16* whh0_16 = (_Float16*)take((size_t)kG4 * kH * 2);
  _Float16* wih1_16 = (_Float16*)take((size_t)kG4 * kH * 2);
  _Float16* whh1_16 = (_Float16*)take((size_t)kG4 * kH * 2);
  _Float16* qw16 = (_Float16*)take((size_t)kH * kH * 2);
  _Float16* kw16 = (_Float16*)take((size_t)kH * kH * 2);
  _Float16* vw16 = (_Float16*)take((size_t)kH * kH * 2);
  _Float16* fc1w16 = (_Float16*)take((size_t)kH * 2 * kH * 2);
  _Float16* fc2w16 = (_Float16*)take((size_t)kNcls * kH * 2);
  float*    xp     = (float*)take((size_t)kML * kG4 * 4);     // reused post-recurrence:
  _Float16* k16    = (_Float16*)xp;                           //  32768*512 f16
  _Float16* v16    = (_Float16*)((char*)xp + 33554432);
  _Float16* vT16   = (_Float16*)((char*)xp + 67108864);
  _Float16* cat16  = (_Float16*)((char*)xp + 100663296);      //  16416*1024 f16
  float*    outs32 = (float*)take((size_t)kML * kH * 4);
  _Float16* h16a = (_Float16*)take((size_t)kB * kH * 2);
  _Float16* h16b = (_Float16*)take((size_t)kB * kH * 2);
  float*    cst  = (float*)take((size_t)kB * kH * 4);
  _Float16* q16  = (_Float16*)take((size_t)kML * kH * 2);
  _Float16* enc16 = (_Float16*)take((size_t)kMS * kH * 2);
  float*    ctx32 = (float*)take((size_t)kML * kH * 4);
  _Float16* h1_16 = (_Float16*)take((size_t)kML * kH * 2);

  // ---- weight & input casts ----
  k_cast16<<<2048, 256, 0, stream>>>(w_ih0, wih0_16, kG4 * kH);
  k_cast16<<<2048, 256, 0, stream>>>(w_hh0, whh0_16, kG4 * kH);
  k_cast16<<<2048, 256, 0, stream>>>(w_ih1, wih1_16, kG4 * kH);
  k_cast16<<<2048, 256, 0, stream>>>(w_hh1, whh1_16, kG4 * kH);
  k_cast16<<<1024, 256, 0, stream>>>(q_w, qw16, kH * kH);
  k_cast16<<<1024, 256, 0, stream>>>(k_w, kw16, kH * kH);
  k_cast16<<<1024, 256, 0, stream>>>(v_w, vw16, kH * kH);
  k_cast16<<<2048, 256, 0, stream>>>(fc1_w, fc1w16, kH * 2 * kH);
  k_cast16<<<512, 256, 0, stream>>>(fc2_w, fc2w16, kNcls * kH);
  k_cast16<<<4096, 256, 0, stream>>>(enc, enc16, kMS * kH);
  k_embed<<<kML, 256, 0, stream>>>(targets, emb_w, emb32, seq16);

  // ---- LSTM layer 0 ----
  k_gemm<kG4, kH, 0><<<dim3(kG4 / 64, kML / 32), 32, 0, stream>>>(seq16, wih0_16, b_ih0,
                                                                  b_hh0, xp, nullptr);
  k_zero16<<<64, 256, 0, stream>>>(h16a, kB * kH);
  k_zero32<<<64, 256, 0, stream>>>(cst, kB * kH);
  for (int t = 0; t < kL; ++t) {
    const _Float16* hin = (t & 1) ? h16b : h16a;
    _Float16* hout = (t & 1) ? h16a : h16b;
    k_lstm_step<<<32, 256, 0, stream>>>(xp, t, whh0_16, hin, hout, cst, nullptr, seq16);
  }
  // ---- LSTM layer 1 ----
  k_gemm<kG4, kH, 0><<<dim3(kG4 / 64, kML / 32), 32, 0, stream>>>(seq16, wih1_16, b_ih1,
                                                                  b_hh1, xp, nullptr);
  k_zero16<<<64, 256, 0, stream>>>(h16a, kB * kH);
  k_zero32<<<64, 256, 0, stream>>>(cst, kB * kH);
  for (int t = 0; t < kL; ++t) {
    const _Float16* hin = (t & 1) ? h16b : h16a;
    _Float16* hout = (t & 1) ? h16a : h16b;
    k_lstm_step<<<32, 256, 0, stream>>>(xp, t, whh1_16, hin, hout, cst, outs32, seq16);
  }

  // ---- attention ----
  k_gemm<kH, kH, 0><<<dim3(kH / 64, kML / 32), 32, 0, stream>>>(seq16, qw16, q_b, nullptr,
                                                                nullptr, q16);
  k_gemm<kH, kH, 0><<<dim3(kH / 64, kMS / 32), 32, 0, stream>>>(enc16, kw16, k_b, nullptr,
                                                                nullptr, k16);
  k_gemm<kH, kH, 0><<<dim3(kH / 64, kMS / 32), 32, 0, stream>>>(enc16, vw16, v_b, nullptr,
                                                                nullptr, v16);
  k_scores<<<dim3(kS / 64, kMT, kB * 4), 32, 0, stream>>>(q16, k16, out_attn);
  k_softmax1024<<<kB * 4 * kL, 256, 0, stream>>>(out_attn);
  k_vtrans<<<4096, 256, 0, stream>>>(v16, vT16);
  k_ctx<<<dim3(kDh / 64, kMT, kB * 4), 32, 0, stream>>>(out_attn, vT16, ctx32);

  // ---- head: fc1(tanh) -> fc2 -> log_softmax ----
  k_cat<<<4096, 256, 0, stream>>>(outs32, ctx32, cat16);
  k_gemm<kH, 2 * kH, 1><<<dim3(kH / 64, kML / 32), 32, 0, stream>>>(cat16, fc1w16, fc1_b,
                                                                    nullptr, nullptr, h1_16);
  k_gemm<kNcls, kH, 0><<<dim3(kNcls / 64, kML / 32), 32, 0, stream>>>(h1_16, fc2w16, fc2_b,
                                                                      nullptr, out_logp, nullptr);
  k_logsoftmax<<<kML, 128, 0, stream>>>(out_logp);

  // ---- word split scatter ----
  k_zero32<<<4096, 256, 0, stream>>>(out_mis, kTW * 7 * 1024);
  k_split<<<kB, 256, 0, stream>>>(targets, emb32, ctx32, out_mis);
}
